// Head_21414706937895
// MI455X (gfx1250) — matrix operand order repaired and checked by reading, not compiled, not run on hardware
//
#include <hip/hip_runtime.h>
#include <hip/hip_bf16.h>
#include <stdint.h>

#define N_EMBD 1024
#define HEAD   64
#define WINDOW 64
#define NGLOB  64
#define NRAND  64
#define BB     8
#define TT     2048

typedef __attribute__((ext_vector_type(16))) __bf16 v16bf;
typedef __attribute__((ext_vector_type(8)))  __bf16 v8bf;
typedef __attribute__((ext_vector_type(8)))  float  v8f;

static __device__ __forceinline__ v8f wmma_bf16(v16bf a, v16bf b, v8f c) {
  return __builtin_amdgcn_wmma_f32_16x16x32_bf16(false, a, false, b, (short)0, c,
                                                 false, false);
}

// Async DMA of one 16-byte chunk global -> LDS (gfx1250, ASYNCcnt-tracked).
static __device__ __forceinline__ void async_copy16(unsigned lds_off,
                                                    const void* gaddr) {
  asm volatile("global_load_async_to_lds_b128 %0, %1, off"
               :
               : "v"(lds_off), "v"(gaddr)
               : "memory");
}

// Convert 16 consecutive fp32 to a v16bf (B-tile: lane holds K=0..15 contiguous).
static __device__ __forceinline__ v16bf cvt16(const float* __restrict__ p) {
  v16bf r;
#pragma unroll
  for (int i = 0; i < 16; i += 4) {
    float4 f = *(const float4*)(p + i);
    r[i + 0] = (__bf16)f.x; r[i + 1] = (__bf16)f.y;
    r[i + 2] = (__bf16)f.z; r[i + 3] = (__bf16)f.w;
  }
  return r;
}

// A-tile from fp32: elements 0..7 = p[0..7], elements 8..15 = p[16..23]
// (caller passes p = row + k0 + half*8; see 16-bit A 16x32 layout).
static __device__ __forceinline__ v16bf cvtA(const float* __restrict__ p) {
  v16bf r;
  float4 f0 = *(const float4*)(p + 0);
  float4 f1 = *(const float4*)(p + 4);
  float4 f2 = *(const float4*)(p + 16);
  float4 f3 = *(const float4*)(p + 20);
  r[0] = (__bf16)f0.x;  r[1] = (__bf16)f0.y;  r[2] = (__bf16)f0.z;  r[3] = (__bf16)f0.w;
  r[4] = (__bf16)f1.x;  r[5] = (__bf16)f1.y;  r[6] = (__bf16)f1.z;  r[7] = (__bf16)f1.w;
  r[8] = (__bf16)f2.x;  r[9] = (__bf16)f2.y;  r[10] = (__bf16)f2.z; r[11] = (__bf16)f2.w;
  r[12] = (__bf16)f3.x; r[13] = (__bf16)f3.y; r[14] = (__bf16)f3.z; r[15] = (__bf16)f3.w;
  return r;
}

// ---------------------------------------------------------------------------
// Kernel 1: per-row mask bitmap: (local | global | random) & causal.
// One thread per row; row fully owned -> no atomics.
// ---------------------------------------------------------------------------
__global__ __launch_bounds__(256) void mask_kernel(
    const int* __restrict__ random_cols, unsigned* __restrict__ maskw) {
  const int row = blockIdx.x * blockDim.x + threadIdx.x;
  if (row >= TT) return;
  const int NW = TT / 32;  // 64 words per row
  for (int w = 0; w < NW; ++w) {
    unsigned bits = 0u;
    const int cbase = w * 32;
    for (int j = 0; j < 32; ++j) {
      const int col = cbase + j;
      const bool causal = (col <= row);
      const bool local = (col >= row - (WINDOW - 1)) && (col <= row);
      const bool glob = (row < NGLOB) || (col < NGLOB);
      if ((local || glob) && causal) bits |= (1u << j);
    }
    maskw[(size_t)row * NW + w] = bits;
  }
  for (int j = 0; j < NRAND; ++j) {
    const int c = random_cols[(size_t)row * NRAND + j];
    if (c <= row)  // causal-AND applied here
      maskw[(size_t)row * NW + (c >> 5)] |= (1u << (c & 31));
  }
}

// ---------------------------------------------------------------------------
// Kernel 2: fused QKV projection.  C[M=B*T, 64] = X[M,1024] * W^T, three W's.
// One wave owns 16 rows; A (x tile) loaded once per K-step, reused for
// 12 WMMA n-tiles (Q,K,V x 4).  Outputs: Q,K row-major bf16, V transposed.
// ---------------------------------------------------------------------------
__global__ __launch_bounds__(256) void proj_kernel(
    const float* __restrict__ x, const float* __restrict__ Wq,
    const float* __restrict__ Wk, const float* __restrict__ Wv,
    __bf16* __restrict__ Qb, __bf16* __restrict__ Kb, __bf16* __restrict__ Vt) {
  const int lane = threadIdx.x & 31;
  const int half = lane >> 4;
  const int mr = lane & 15;
  const int gw = blockIdx.x * (blockDim.x >> 5) + (threadIdx.x >> 5);
  const int row0 = gw * 16;

  v8f accQ[4] = {}, accK[4] = {}, accV[4] = {};
  const float* xrow = x + (size_t)(row0 + mr) * N_EMBD;

  for (int k0 = 0; k0 < N_EMBD; k0 += 32) {
    const v16bf A = cvt16(xrow + k0 + half * 16);
#pragma unroll
    for (int nt = 0; nt < 4; ++nt) {
      const size_t boff = (size_t)(nt * 16 + mr) * N_EMBD + k0 + half * 16;
      accQ[nt] = wmma_bf16(A, cvt16(Wq + boff), accQ[nt]);
      accK[nt] = wmma_bf16(A, cvt16(Wk + boff), accK[nt]);
      accV[nt] = wmma_bf16(A, cvt16(Wv + boff), accV[nt]);
    }
  }

  const int bidx = row0 / TT;
  const int t0 = row0 % TT;
#pragma unroll
  for (int nt = 0; nt < 4; ++nt) {
    const int h = nt * 16 + mr;
#pragma unroll
    for (int r = 0; r < 8; ++r) {
      const int m = half * 8 + r;  // C-layout: VGPR r -> M = half*8 + r
      Qb[(size_t)(row0 + m) * HEAD + h] = (__bf16)accQ[nt][r];
      Kb[(size_t)(row0 + m) * HEAD + h] = (__bf16)accK[nt][r];
      Vt[((size_t)bidx * HEAD + h) * TT + t0 + m] = (__bf16)accV[nt][r];
    }
  }
}

// ---------------------------------------------------------------------------
// Kernel 3: dense-causal flash attention with bitmask.  One workgroup per
// (batch, 64-row query block); 4 waves, each owning a 16-row strip.
// K/V blocks staged global->LDS once per workgroup via async DMA,
// double-buffered (ASYNCcnt); WMMA B-operands then come from LDS.
// ---------------------------------------------------------------------------
__global__ __launch_bounds__(128) void attn_kernel(
    const __bf16* __restrict__ Qb, const __bf16* __restrict__ Kb,
    const __bf16* __restrict__ Vt, const unsigned* __restrict__ maskw,
    float* __restrict__ out) {
  __shared__ unsigned smask[64 * 2];
  __shared__ __align__(32) __bf16 sP[4 * 16 * 64];
  __shared__ __align__(32) __bf16 sK[2][64 * 64];  // [n][h], linear copy
  __shared__ __align__(32) __bf16 sV[2][64 * 64];  // [h][n_local]

  const int lane = threadIdx.x & 31;
  const int half = lane >> 4;
  const int mr = lane & 15;
  const int wv = threadIdx.x >> 5;  // strip 0..3
  const int qb = blockIdx.x;
  const int b = blockIdx.y;
  const int q0 = qb * 64;

  const __bf16* kbase = Kb + (size_t)b * TT * HEAD;
  const __bf16* vbase = Vt + (size_t)b * HEAD * TT;

  // Cooperative async stage of one 64x64 K block + V block into buffer `buf`.
  auto issue_async = [&](int kb, int buf) {
    const char* kg = (const char*)(kbase + (size_t)kb * 64 * HEAD);
    const unsigned kl = (unsigned)(uintptr_t)(&sK[buf][0]);
    const unsigned vl = (unsigned)(uintptr_t)(&sV[buf][0]);
#pragma unroll
    for (int i = 0; i < 4; ++i) {
      const int idx = i * 128 + threadIdx.x;  // 16B chunk id, 0..511
      async_copy16(kl + idx * 16, kg + idx * 16);  // K block is contiguous
      const int vrow = idx >> 3, vcol = idx & 7;   // V^T rows stride T
      const char* vg =
          (const char*)vbase + ((size_t)vrow * TT + kb * 64) * 2 + vcol * 16;
      async_copy16(vl + idx * 16, vg);
    }
  };

  // Q strip in A layout (K = head dim, two 32-chunks)
  const __bf16* pq = Qb + ((size_t)b * TT + q0 + wv * 16 + mr) * HEAD;
  v16bf qa[2];
#pragma unroll
  for (int c = 0; c < 2; ++c) {
    v8bf lo = *(const v8bf*)(pq + c * 32 + half * 16);
    v8bf hi = *(const v8bf*)(pq + c * 32 + half * 16 + 8);
    qa[c] = __builtin_shufflevector(lo, hi, 0, 1, 2, 3, 4, 5, 6, 7, 8, 9, 10,
                                    11, 12, 13, 14, 15);
  }

  float mx[8], l[8];
  v8f o[4] = {};
#pragma unroll
  for (int r = 0; r < 8; ++r) { mx[r] = -__builtin_inff(); l[r] = 0.f; }

  issue_async(0, 0);  // prologue: stream first K/V block

  for (int kb = 0; kb <= qb; ++kb) {
    const int buf = kb & 1;
    asm volatile("s_wait_asynccnt 0" ::: "memory");  // own DMA chunks done
    __syncthreads();  // all waves' DMA done; prior smask/sK/sV reads done

    {
      const int r = threadIdx.x >> 1, w = threadIdx.x & 1;
      smask[r * 2 + w] = maskw[(size_t)(q0 + r) * (TT / 32) + kb * 2 + w];
    }
    if (kb < qb) issue_async(kb + 1, buf ^ 1);  // overlap next block's DMA
    if (kb + 2 <= qb)  // warm L2 ahead of the async engine
      __builtin_prefetch(kbase + (size_t)(kb + 2) * 64 * HEAD, 0, 0);
    __syncthreads();

    // S = Q * K^T for this strip (4 n-tiles x 2 k-chunks), K from LDS
    v8f s[4];
#pragma unroll
    for (int nt = 0; nt < 4; ++nt) {
      v8f acc = {};
#pragma unroll
      for (int c = 0; c < 2; ++c) {
        const __bf16* pk = &sK[buf][(nt * 16 + mr) * 64 + c * 32 + half * 16];
        acc = wmma_bf16(qa[c], *(const v16bf*)pk, acc);
      }
      s[nt] = acc;
    }

    // mask + scale (1/sqrt(64))
#pragma unroll
    for (int nt = 0; nt < 4; ++nt) {
      const int ccol = nt * 16 + mr;
#pragma unroll
      for (int r = 0; r < 8; ++r) {
        const int rrow = wv * 16 + half * 8 + r;
        const unsigned bits = smask[rrow * 2 + (ccol >> 5)];
        s[nt][r] = ((bits >> (ccol & 31)) & 1u) ? s[nt][r] * 0.125f
                                                : -__builtin_inff();
      }
    }

    // online softmax: rows live in 16-lane halves -> xor-tree over {8,4,2,1}
    float corr[8];
#pragma unroll
    for (int r = 0; r < 8; ++r) {
      float v = fmaxf(fmaxf(s[0][r], s[1][r]), fmaxf(s[2][r], s[3][r]));
#pragma unroll
      for (int off = 8; off >= 1; off >>= 1) v = fmaxf(v, __shfl_xor(v, off, 32));
      const float nm = fmaxf(mx[r], v);
      corr[r] = __expf(mx[r] - nm);
      mx[r] = nm;
      l[r] *= corr[r];
    }
#pragma unroll
    for (int nt = 0; nt < 4; ++nt)
#pragma unroll
      for (int r = 0; r < 8; ++r) {
        s[nt][r] = __expf(s[nt][r] - mx[r]);
        o[nt][r] *= corr[r];
      }
#pragma unroll
    for (int r = 0; r < 8; ++r) {
      float v = (s[0][r] + s[1][r]) + (s[2][r] + s[3][r]);
#pragma unroll
      for (int off = 8; off >= 1; off >>= 1) v += __shfl_xor(v, off, 32);
      l[r] += v;
    }

    // P: C-layout -> bf16 row-major in per-wave LDS -> reread in A layout
    __bf16* pp = sP + wv * 16 * 64;
#pragma unroll
    for (int nt = 0; nt < 4; ++nt)
#pragma unroll
      for (int r = 0; r < 8; ++r)
        pp[(half * 8 + r) * 64 + nt * 16 + mr] = (__bf16)s[nt][r];
    asm volatile("s_wait_dscnt 0" ::: "memory");

    v16bf pa[2];
#pragma unroll
    for (int c = 0; c < 2; ++c) {
      v8bf lo = *(const v8bf*)(pp + mr * 64 + c * 32 + half * 16);
      v8bf hi = *(const v8bf*)(pp + mr * 64 + c * 32 + half * 16 + 8);
      pa[c] = __builtin_shufflevector(lo, hi, 0, 1, 2, 3, 4, 5, 6, 7, 8, 9, 10,
                                      11, 12, 13, 14, 15);
    }

    // O += P * V, V from LDS (V^T layout -> contiguous B-operand reads)
#pragma unroll
    for (int nt = 0; nt < 4; ++nt)
#pragma unroll
      for (int c = 0; c < 2; ++c) {
        const __bf16* pv = &sV[buf][(nt * 16 + mr) * 64 + c * 32 + half * 16];
        o[nt] = wmma_bf16(pa[c], *(const v16bf*)pv, o[nt]);
      }
  }

#pragma unroll
  for (int nt = 0; nt < 4; ++nt)
#pragma unroll
    for (int r = 0; r < 8; ++r) {
      const int rrow = q0 + wv * 16 + half * 8 + r;
      out[((size_t)b * TT + rrow) * HEAD + nt * 16 + mr] = o[nt][r] / l[r];
    }
}

// ---------------------------------------------------------------------------
extern "C" void kernel_launch(void* const* d_in, const int* in_sizes, int n_in,
                              void* d_out, int out_size, void* d_ws,
                              size_t ws_size, hipStream_t stream) {
  const float* x = (const float*)d_in[0];
  const int* rcols = (const int*)d_in[1];
  const float* Wk = (const float*)d_in[2];
  const float* Wq = (const float*)d_in[3];
  const float* Wv = (const float*)d_in[4];
  float* out = (float*)d_out;

  char* ws = (char*)d_ws;
  const size_t qkbytes = (size_t)BB * TT * HEAD * sizeof(__bf16);  // 2 MB each
  __bf16* Qb = (__bf16*)(ws);
  __bf16* Kb = (__bf16*)(ws + qkbytes);
  __bf16* Vt = (__bf16*)(ws + 2 * qkbytes);
  unsigned* maskw = (unsigned*)(ws + 3 * qkbytes);  // 512 KB

  mask_kernel<<<dim3((TT + 255) / 256), dim3(256), 0, stream>>>(rcols, maskw);
  proj_kernel<<<dim3((BB * TT) / (16 * 8)), dim3(256), 0, stream>>>(
      x, Wq, Wk, Wv, Qb, Kb, Vt);
  attn_kernel<<<dim3(TT / 64, BB), dim3(128), 0, stream>>>(Qb, Kb, Vt, maskw,
                                                           out);
}
